// Hybrid_Model_20255065768517
// MI455X (gfx1250) — matrix operand order repaired and checked by reading, not compiled, not run on hardware
//
#include <hip/hip_runtime.h>
#include <hip/hip_bf16.h>

#define N_NODES 50000
#define N_EDGES 800000
#define F_IN 128
#define HID 128
#define F_EDGE 16
#define F_TAB 16
#define N_CLASSES 3

typedef __attribute__((ext_vector_type(16))) __bf16 v16bf;
typedef __attribute__((ext_vector_type(8)))  __bf16 v8bf;
typedef __attribute__((ext_vector_type(8)))  float  v8f;
typedef unsigned short u16;
typedef int v4i __attribute__((vector_size(4 * sizeof(int))));

// ---------------------------------------------------------------------------
// WMMA fragment helpers (CDNA5 wave32, V_WMMA_F32_16X16X32_BF16)
// A (16x32, MxK): m = lane&15, half = lane>>4,
//   element e -> k = half*16 + e   => two contiguous 8-elt runs
// B (32x16, KxN): n = lane&15, half = lane>>4, element e -> k = half*16 + e
// C/D (16x16 f32): n = lane&15, VGPR v -> m = v + 8*(lane>>4)
// ---------------------------------------------------------------------------
__device__ __forceinline__ v16bf load_a_lds(const __bf16* tile, int ldk, int kc, int lane) {
  const int m = lane & 15, half = lane >> 4;
  const __bf16* p = tile + m * ldk + kc + half * 16;
  v8bf lo = *(const v8bf*)(p);
  v8bf hi = *(const v8bf*)(p + 8);
  v16bf a;
#pragma unroll
  for (int i = 0; i < 8; ++i) { a[i] = lo[i]; a[8 + i] = hi[i]; }
  return a;
}

__device__ __forceinline__ v16bf load_b_glb(const u16* Wt_, int K, int nbase, int kc, int lane) {
  const __bf16* Wt = (const __bf16*)Wt_;
  const int n = nbase + (lane & 15), half = lane >> 4;
  return *(const v16bf*)(Wt + (size_t)n * K + kc + half * 16);
}

__device__ __forceinline__ v8f wmma_bf16(v16bf a, v16bf b, v8f c) {
  return __builtin_amdgcn_wmma_f32_16x16x32_bf16(false, a, false, b, (short)0, c, false, false);
}

// One K-chunk across 4 M-tiles: preload all A fragments, then issue the WMMA
// quartet so the ds_loads can be clause'd and waited with partial dscnt.
__device__ __forceinline__ void wmma_step4(const __bf16* tile, int tileStride, int ldk,
                                           int kc, int lane, v16bf b, v8f acc[4]) {
  v16bf a0 = load_a_lds(tile + 0 * tileStride, ldk, kc, lane);
  v16bf a1 = load_a_lds(tile + 1 * tileStride, ldk, kc, lane);
  v16bf a2 = load_a_lds(tile + 2 * tileStride, ldk, kc, lane);
  v16bf a3 = load_a_lds(tile + 3 * tileStride, ldk, kc, lane);
  acc[0] = wmma_bf16(a0, b, acc[0]);
  acc[1] = wmma_bf16(a1, b, acc[1]);
  acc[2] = wmma_bf16(a2, b, acc[2]);
  acc[3] = wmma_bf16(a3, b, acc[3]);
}

__device__ __forceinline__ void wait_async_lds() {
#if __has_builtin(__builtin_amdgcn_s_wait_asynccnt)
  __builtin_amdgcn_s_wait_asynccnt(0);
#else
  asm volatile("s_wait_asynccnt 0x0" ::: "memory");
#endif
}

// ---------------------------------------------------------------------------
// Utility kernels
// ---------------------------------------------------------------------------
__global__ void k_zero_f32(float* p, int n) {
  int i = blockIdx.x * blockDim.x + threadIdx.x;
  if (i < n) p[i] = 0.0f;
}

__global__ void k_count(const int* dst, float* cnt) {
  int e = blockIdx.x * blockDim.x + threadIdx.x;
  if (e < N_EDGES) atomicAdd(&cnt[dst[e]], 1.0f);
}

__global__ void k_invcnt(const float* cnt, float* invc) {
  int i = blockIdx.x * blockDim.x + threadIdx.x;
  if (i < N_NODES) invc[i] = 1.0f / fmaxf(cnt[i], 1.0f);
}

// agg[dst[e]] += x[src[e]]  (fp32 source), 4 features per thread
__global__ void k_scatter_f32(const float* x, const int* src, const int* dst, float* agg) {
  int idx = blockIdx.x * blockDim.x + threadIdx.x;
  if (idx >= N_EDGES * 32) return;
  int e = idx >> 5, j = (idx & 31) * 4;
  int s = src[e], d = dst[e];
  const float4 v = *(const float4*)(x + (size_t)s * 128 + j);
  atomicAdd(&agg[(size_t)d * 128 + j + 0], v.x);
  atomicAdd(&agg[(size_t)d * 128 + j + 1], v.y);
  atomicAdd(&agg[(size_t)d * 128 + j + 2], v.z);
  atomicAdd(&agg[(size_t)d * 128 + j + 3], v.w);
}

// agg[dst[e]] += h[src[e]]  (bf16 source)
__global__ void k_scatter_bf16(const u16* h_, const int* src, const int* dst, float* agg) {
  const __bf16* h = (const __bf16*)h_;
  int idx = blockIdx.x * blockDim.x + threadIdx.x;
  if (idx >= N_EDGES * 32) return;
  int e = idx >> 5, j = (idx & 31) * 4;
  int s = src[e], d = dst[e];
  const __bf16* p = h + (size_t)s * 128 + j;
#pragma unroll
  for (int t = 0; t < 4; ++t)
    atomicAdd(&agg[(size_t)d * 128 + j + t], (float)p[t]);
}

// Build transposed bf16 weight packs + folded BN scale/shift.
// W1t/W2t: [N=128][K=256] with K = [Wl rows | Wr rows]
// Wc1t: [N=128][K=288], Wc2t: [N=64][K=128]
__global__ void k_prep(const float* W1l, const float* W1r,
                       const float* W2l, const float* W2r,
                       const float* Wc1, const float* Wc2,
                       const float* b1l, const float* g1, const float* be1,
                       const float* m1, const float* v1,
                       const float* b2l, const float* g2, const float* be2,
                       const float* m2, const float* v2,
                       u16* W1t_, u16* W2t_, u16* Wc1t_, u16* Wc2t_,
                       float* bn1s, float* bn1b, float* bn2s, float* bn2b) {
  __bf16* W1t = (__bf16*)W1t_;  __bf16* W2t = (__bf16*)W2t_;
  __bf16* Wc1t = (__bf16*)Wc1t_; __bf16* Wc2t = (__bf16*)Wc2t_;
  int idx = blockIdx.x * blockDim.x + threadIdx.x;
  if (idx < 128 * 256) {
    int n = idx >> 8, k = idx & 255;
    W1t[n * 256 + k] = (__bf16)((k < 128) ? W1l[k * 128 + n] : W1r[(k - 128) * 128 + n]);
    W2t[n * 256 + k] = (__bf16)((k < 128) ? W2l[k * 128 + n] : W2r[(k - 128) * 128 + n]);
  }
  if (idx < 128 * 288) {
    int n = idx / 288, k = idx - n * 288;
    Wc1t[n * 288 + k] = (__bf16)Wc1[k * 128 + n];
  }
  if (idx < 64 * 128) {
    int n = idx >> 7, k = idx & 127;
    Wc2t[n * 128 + k] = (__bf16)Wc2[k * 64 + n];
  }
  if (idx < 128) {
    float s1 = g1[idx] * rsqrtf(v1[idx] + 1e-5f);
    bn1s[idx] = s1;
    bn1b[idx] = (b1l[idx] - m1[idx]) * s1 + be1[idx];
    float s2 = g2[idx] * rsqrtf(v2[idx] + 1e-5f);
    bn2s[idx] = s2;
    bn2b[idx] = (b2l[idx] - m2[idx]) * s2 + be2[idx];
  }
}

// ---------------------------------------------------------------------------
// Node layer: h = relu(bn(agg_mean @ Wl + bl + x @ Wr))
// 64 rows (4 M-tiles) per block; 8 waves cover N=128, each wave keeps 4
// independent accumulator chains and reuses every B fragment 4x.
// ---------------------------------------------------------------------------
template <bool XBF16>
__global__ void k_node_gemm(const float* agg, const float* invc,
                            const float* xf, const u16* xb_,
                            const u16* Wt, const float* scale, const float* shift,
                            u16* out_) {
  __shared__ alignas(16) __bf16 tileA[64 * 256];   // 32 KB
  const __bf16* xb = (const __bf16*)xb_;
  __bf16* out = (__bf16*)out_;
  const int row0 = blockIdx.x * 64;
  const int tid = threadIdx.x;

  // cols 0..127 : agg * invc (f32 -> bf16), 4 floats per iteration
  for (int idx = tid; idx < 64 * 32; idx += 256) {
    int r = idx >> 5, cc = idx & 31;
    int node = row0 + r;
    float4 v = {0.f, 0.f, 0.f, 0.f};
    if (node < N_NODES) {
      v = *(const float4*)(agg + (size_t)node * 128 + cc * 4);
      float ic = invc[node];
      v.x *= ic; v.y *= ic; v.z *= ic; v.w *= ic;
    }
    __bf16* d = &tileA[r * 256 + cc * 4];
    d[0] = (__bf16)v.x; d[1] = (__bf16)v.y; d[2] = (__bf16)v.z; d[3] = (__bf16)v.w;
  }
  // cols 128..255 : previous features
  if (XBF16) {
    for (int idx = tid; idx < 64 * 16; idx += 256) {
      int r = idx >> 4, cc = idx & 15;
      int node = row0 + r;
      v8bf v = {};
      if (node < N_NODES) v = *(const v8bf*)(xb + (size_t)node * 128 + cc * 8);
      *(v8bf*)(&tileA[r * 256 + 128 + cc * 8]) = v;
    }
  } else {
    for (int idx = tid; idx < 64 * 32; idx += 256) {
      int r = idx >> 5, cc = idx & 31;
      int node = row0 + r;
      float4 v = {0.f, 0.f, 0.f, 0.f};
      if (node < N_NODES) v = *(const float4*)(xf + (size_t)node * 128 + cc * 4);
      __bf16* d = &tileA[r * 256 + 128 + cc * 4];
      d[0] = (__bf16)v.x; d[1] = (__bf16)v.y; d[2] = (__bf16)v.z; d[3] = (__bf16)v.w;
    }
  }
  __syncthreads();

  const int wave = tid >> 5, lane = tid & 31;
  const int nbase = wave * 16;
  v8f acc[4] = {{}, {}, {}, {}};
#pragma unroll
  for (int kc = 0; kc < 256; kc += 32) {
    v16bf b = load_b_glb(Wt, 256, nbase, kc, lane);
    wmma_step4(tileA, 16 * 256, 256, kc, lane, b, acc);
  }
  const int n = nbase + (lane & 15), half = lane >> 4;
  const float s = scale[n], t = shift[n];
#pragma unroll
  for (int mt = 0; mt < 4; ++mt) {
#pragma unroll
    for (int v = 0; v < 8; ++v) {
      int node = row0 + mt * 16 + v + half * 8;
      if (node < N_NODES) {
        float y = acc[mt][v] * s + t;
        y = y > 0.0f ? y : 0.0f;
        out[(size_t)node * 128 + n] = (__bf16)y;
      }
    }
  }
}

// ---------------------------------------------------------------------------
// Fused edge classifier: 64 edges (4 M-tiles) per block.  N_EDGES % 64 == 0,
// so no bounds guards anywhere (EXEC stays all-ones).
// shA[64x288] = [h2[src] | h2[dst] | attr | tab] (bf16)   -- aliased with z2
//  -> z1 = relu(shA @ Wc1 + bc1) [64x128] bf16  (8 waves x 4 tiles, 9 K-chunks)
//  -> z2 = relu(z1 @ Wc2 + bc2)  [64x64]  f32   (4 waves x 4 tiles, 4 K-chunks)
//  -> out = z2 @ Wc3 + bc3       [64x3]   VALU
// ---------------------------------------------------------------------------
__global__ void k_edge_mlp(const int* src, const int* dst,
                           const u16* h2_, const float* eattr, const float* tab,
                           const u16* Wc1t, const float* bc1,
                           const u16* Wc2t, const float* bc2,
                           const float* Wc3, const float* bc3,
                           float* out) {
  __shared__ alignas(16) unsigned char smem[64 * 288 * 2];  // 36 KB: shA / z2
  __shared__ alignas(16) __bf16 z1[64 * 128];               // 16 KB
  __bf16* shA = (__bf16*)smem;
  float*  z2  = (float*)smem;   // alive only after all shA reads complete
  const __bf16* h2 = (const __bf16*)h2_;
  const int e0 = blockIdx.x * 64;
  const int tid = threadIdx.x;

  // cols 0..255 : h2[src] | h2[dst], 16 B per lane per iteration. Whole wave
  // shares a row so src[e]/dst[e] become scalar loads.
#if __has_builtin(__builtin_amdgcn_global_load_async_to_lds_b128)
  // gfx1250 async DMA path: global -> LDS without VGPR round-trip (ASYNCcnt).
  for (int idx = tid; idx < 64 * 32; idx += 256) {
    int r = idx >> 5, cc = idx & 31;
    int e = e0 + r;
    int node = (cc < 16) ? src[e] : dst[e];
    __builtin_amdgcn_global_load_async_to_lds_b128(
        (v4i*)(h2 + (size_t)node * 128 + (cc & 15) * 8),
        (v4i*)(&shA[r * 288 + cc * 8]), 0, 0);
  }
#else
  for (int idx = tid; idx < 64 * 32; idx += 256) {
    int r = idx >> 5, cc = idx & 31;
    int e = e0 + r;
    int node = (cc < 16) ? src[e] : dst[e];
    v8bf v = *(const v8bf*)(h2 + (size_t)node * 128 + (cc & 15) * 8);
    *(v8bf*)(&shA[r * 288 + cc * 8]) = v;
  }
#endif
  // cols 256..287 : edge_attr | tabular (f32 -> bf16), overlaps the async DMA
  for (int idx = tid; idx < 64 * 32; idx += 256) {
    int r = idx >> 5, c = idx & 31;
    int e = e0 + r;
    float v = (c < 16) ? eattr[(size_t)e * 16 + c] : tab[(size_t)e * 16 + (c - 16)];
    shA[r * 288 + 256 + c] = (__bf16)v;
  }
#if __has_builtin(__builtin_amdgcn_global_load_async_to_lds_b128)
  wait_async_lds();
#endif
  __syncthreads();

  const int wave = tid >> 5, lane = tid & 31;
  {  // GEMM1: 288 -> 128
    const int nbase = wave * 16;
    v8f acc[4] = {{}, {}, {}, {}};
#pragma unroll
    for (int kc = 0; kc < 288; kc += 32) {
      v16bf b = load_b_glb(Wc1t, 288, nbase, kc, lane);
      wmma_step4(shA, 16 * 288, 288, kc, lane, b, acc);
    }
    const int n = nbase + (lane & 15), half = lane >> 4;
    const float bb = bc1[n];
#pragma unroll
    for (int mt = 0; mt < 4; ++mt) {
#pragma unroll
      for (int v = 0; v < 8; ++v) {
        float y = acc[mt][v] + bb;
        y = y > 0.0f ? y : 0.0f;
        z1[(mt * 16 + v + half * 8) * 128 + n] = (__bf16)y;
      }
    }
  }
  __syncthreads();   // shA dead past this point; z2 may overlay it

  if (wave < 4) {  // GEMM2: 128 -> 64
    const int nbase = wave * 16;
    v8f acc[4] = {{}, {}, {}, {}};
#pragma unroll
    for (int kc = 0; kc < 128; kc += 32) {
      v16bf b = load_b_glb(Wc2t, 128, nbase, kc, lane);
      wmma_step4(z1, 16 * 128, 128, kc, lane, b, acc);
    }
    const int n = nbase + (lane & 15), half = lane >> 4;
    const float bb = bc2[n];
#pragma unroll
    for (int mt = 0; mt < 4; ++mt) {
#pragma unroll
      for (int v = 0; v < 8; ++v) {
        float y = acc[mt][v] + bb;
        y = y > 0.0f ? y : 0.0f;
        z2[(mt * 16 + v + half * 8) * 64 + n] = y;
      }
    }
  }
  __syncthreads();

  if (tid < 192) {  // GEMM3: 64 -> 3 (64 edges x 3 classes)
    int r = tid / 3, c = tid - r * 3;
    int e = e0 + r;
    float s = bc3[c];
#pragma unroll
    for (int k = 0; k < 64; ++k) s += z2[r * 64 + k] * Wc3[k * 3 + c];
    out[(size_t)e * 3 + c] = s;
  }
}

// ---------------------------------------------------------------------------
// Host launch
// ---------------------------------------------------------------------------
extern "C" void kernel_launch(void* const* d_in, const int* in_sizes, int n_in,
                              void* d_out, int out_size, void* d_ws, size_t ws_size,
                              hipStream_t stream) {
  (void)in_sizes; (void)n_in; (void)out_size; (void)ws_size;
  const float* x    = (const float*)d_in[0];
  const int*   ei   = (const int*)d_in[1];
  const int*   src  = ei;
  const int*   dst  = ei + N_EDGES;
  const float* attr = (const float*)d_in[2];
  const float* tab  = (const float*)d_in[3];
  const float* W1l  = (const float*)d_in[4];
  const float* b1l  = (const float*)d_in[5];
  const float* W1r  = (const float*)d_in[6];
  const float* W2l  = (const float*)d_in[7];
  const float* b2l  = (const float*)d_in[8];
  const float* W2r  = (const float*)d_in[9];
  const float* g1   = (const float*)d_in[10];
  const float* be1  = (const float*)d_in[11];
  const float* m1   = (const float*)d_in[12];
  const float* v1   = (const float*)d_in[13];
  const float* g2   = (const float*)d_in[14];
  const float* be2  = (const float*)d_in[15];
  const float* m2   = (const float*)d_in[16];
  const float* v2   = (const float*)d_in[17];
  const float* bc1  = (const float*)d_in[19];
  const float* bc2  = (const float*)d_in[21];
  const float* Wc3  = (const float*)d_in[22];
  const float* bc3  = (const float*)d_in[23];
  float* out = (float*)d_out;

  // Carve workspace (256B aligned chunks)
  char* p = (char*)d_ws;
  auto carve = [&](size_t bytes) { void* r = (void*)p; p += (bytes + 255) & ~(size_t)255; return r; };
  float* cnt   = (float*)carve((size_t)N_NODES * 4);
  float* invc  = (float*)carve((size_t)N_NODES * 4);
  float* agg   = (float*)carve((size_t)N_NODES * 128 * 4);
  u16*   h1    = (u16*)carve((size_t)N_NODES * 128 * 2);
  u16*   h2    = (u16*)carve((size_t)N_NODES * 128 * 2);
  u16*   W1t   = (u16*)carve(128 * 256 * 2);
  u16*   W2t   = (u16*)carve(128 * 256 * 2);
  u16*   Wc1t  = (u16*)carve(128 * 288 * 2);
  u16*   Wc2t  = (u16*)carve(64 * 128 * 2);
  float* bn1s  = (float*)carve(128 * 4);
  float* bn1b  = (float*)carve(128 * 4);
  float* bn2s  = (float*)carve(128 * 4);
  float* bn2b  = (float*)carve(128 * 4);

  const int TB = 256;
  const int aggN = N_NODES * 128;
  const int scatN = N_EDGES * 32;

  // Prep: weights/bn folding + degree counts
  k_prep<<<(128 * 288 + TB - 1) / TB, TB, 0, stream>>>(
      W1l, W1r, W2l, W2r, (const float*)d_in[18], (const float*)d_in[20],
      b1l, g1, be1, m1, v1, b2l, g2, be2, m2, v2,
      W1t, W2t, Wc1t, Wc2t, bn1s, bn1b, bn2s, bn2b);
  k_zero_f32<<<(N_NODES + TB - 1) / TB, TB, 0, stream>>>(cnt, N_NODES);
  k_zero_f32<<<(aggN + TB - 1) / TB, TB, 0, stream>>>(agg, aggN);
  k_count<<<(N_EDGES + TB - 1) / TB, TB, 0, stream>>>(dst, cnt);
  k_invcnt<<<(N_NODES + TB - 1) / TB, TB, 0, stream>>>(cnt, invc);

  // Layer 1
  k_scatter_f32<<<(scatN + TB - 1) / TB, TB, 0, stream>>>(x, src, dst, agg);
  k_node_gemm<false><<<(N_NODES + 63) / 64, TB, 0, stream>>>(
      agg, invc, x, (const u16*)nullptr, W1t, bn1s, bn1b, h1);

  // Layer 2
  k_zero_f32<<<(aggN + TB - 1) / TB, TB, 0, stream>>>(agg, aggN);
  k_scatter_bf16<<<(scatN + TB - 1) / TB, TB, 0, stream>>>(h1, src, dst, agg);
  k_node_gemm<true><<<(N_NODES + 63) / 64, TB, 0, stream>>>(
      agg, invc, (const float*)nullptr, h1, W2t, bn2s, bn2b, h2);

  // Fused edge classifier
  k_edge_mlp<<<(N_EDGES + 63) / 64, TB, 0, stream>>>(
      src, dst, h2, attr, tab, Wc1t, bc1, Wc2t, bc2, Wc3, bc3, out);
}
